// GCN_77670188581384
// MI455X (gfx1250) — matrix operand order repaired and checked by reading, compiled, not run on hardware
//
#include <hip/hip_runtime.h>

#define N_NODES   50000
#define N_EDGES   600000
#define HIDDEN    128
#define N_CLASSES 10
#define N_GRAPHS  128
#define TOT_EDGES (N_EDGES + N_NODES)
#define FEAT_TOT  (N_NODES * HIDDEN)

typedef __attribute__((ext_vector_type(16))) __bf16 v16bf;
typedef __attribute__((ext_vector_type(8)))  __bf16 v8bf;
typedef __attribute__((ext_vector_type(8)))  float  v8f;

static constexpr int LDSW = 136;  // padded bf16 stride: 68-dword row stride -> conflict-free B reads

// ---------------- WMMA GEMM: Out[M,128] = A[M,128] @ W[128,128] (bf16 inputs, f32 accum) ----
__global__ void __launch_bounds__(256) gemm_bf16_wmma(
    const float* __restrict__ A, const float* __restrict__ W,
    float* __restrict__ Out, int M)
{
  __shared__ __bf16 Wt[128 * LDSW];          // W transposed: Wt[n][k]
  const int tid = threadIdx.x;
  for (int idx = tid; idx < 128 * 128; idx += 256) {
    int k = idx >> 7, n = idx & 127;
    Wt[n * LDSW + k] = (__bf16)W[idx];
  }
  __syncthreads();

  const int lane   = tid & 31;
  const int wave   = tid >> 5;
  const int tileM  = blockIdx.x * 8 + wave;  // one 16x128 output slab per wave
  if (tileM * 16 >= M) return;

  const int laneLo = lane & 15;
  const int laneHi = lane >> 4;              // 0: lanes 0-15, 1: lanes 16-31
  const int row    = tileM * 16 + laneLo;

  v8f acc[8] = {};
#pragma unroll
  for (int kt = 0; kt < 4; ++kt) {
    // A fragment (16x32 bf16): lanes 0-15 K={kb..kb+7, kb+16..kb+23}, kb=kt*32 (+8 for hi half)
    const int kb = kt * 32 + laneHi * 8;
    const float* ap = A + (size_t)row * 128 + kb;
    const float4 f0 = *(const float4*)(ap + 0);
    const float4 f1 = *(const float4*)(ap + 4);
    const float4 f2 = *(const float4*)(ap + 16);
    const float4 f3 = *(const float4*)(ap + 20);
    v16bf a;
    a[0]=(__bf16)f0.x;  a[1]=(__bf16)f0.y;  a[2]=(__bf16)f0.z;  a[3]=(__bf16)f0.w;
    a[4]=(__bf16)f1.x;  a[5]=(__bf16)f1.y;  a[6]=(__bf16)f1.z;  a[7]=(__bf16)f1.w;
    a[8]=(__bf16)f2.x;  a[9]=(__bf16)f2.y;  a[10]=(__bf16)f2.z; a[11]=(__bf16)f2.w;
    a[12]=(__bf16)f3.x; a[13]=(__bf16)f3.y; a[14]=(__bf16)f3.z; a[15]=(__bf16)f3.w;
#pragma unroll
    for (int nt = 0; nt < 8; ++nt) {
      // B fragment (32x16): lane holds col nt*16+laneLo, K={kb..kb+7, kb+16..kb+23} (kb += 8 for hi half)
      const __bf16* bp = &Wt[(nt * 16 + laneLo) * LDSW + kt * 32 + laneHi * 8];
      v8bf blo = *(const v8bf*)(bp);
      v8bf bhi = *(const v8bf*)(bp + 16);
      v16bf b;
#pragma unroll
      for (int e = 0; e < 8; ++e) { b[e] = blo[e]; b[e + 8] = bhi[e]; }
      acc[nt] = __builtin_amdgcn_wmma_f32_16x16x32_bf16(
          false, a, false, b, (short)0, acc[nt], false, false);
    }
  }
  // C/D layout: VGPR r -> row (laneHi? 8+r : r), col = lane&15 within n-tile
  const int mbase = tileM * 16 + laneHi * 8;
#pragma unroll
  for (int nt = 0; nt < 8; ++nt) {
    const int col = nt * 16 + laneLo;
#pragma unroll
    for (int r = 0; r < 8; ++r)
      Out[(size_t)(mbase + r) * 128 + col] = acc[nt][r];
  }
}

// ---------------- normalization precompute ----------------
__global__ void fill_f32(float* __restrict__ p, float v, int n) {
  int i = blockIdx.x * blockDim.x + threadIdx.x;
  if (i < n) p[i] = v;
}
__global__ void deg_count(const int* __restrict__ ei, float* __restrict__ deg) {
  int e = blockIdx.x * blockDim.x + threadIdx.x;
  if (e < N_EDGES) atomicAdd(&deg[ei[N_EDGES + e]], 1.0f);  // col = edge_index[1]
}
__global__ void deg_to_dinv(float* __restrict__ d, int n) {
  int i = blockIdx.x * blockDim.x + threadIdx.x;
  if (i < n) { float v = d[i]; d[i] = (v > 0.0f) ? rsqrtf(v) : 0.0f; }
}

// ---------------- per-layer elementwise ----------------
__global__ void bias_broadcast(float* __restrict__ agg, const float* __restrict__ b, int total) {
  int i = blockIdx.x * blockDim.x + threadIdx.x;
  if (i < total) agg[i] = b[i & 127];
}
__global__ void relu_inplace(float* __restrict__ a, int total) {
  int i = blockIdx.x * blockDim.x + threadIdx.x;
  if (i < total) a[i] = fmaxf(a[i], 0.0f);
}
__global__ void bias_relu_inplace(float* __restrict__ a, const float* __restrict__ b, int total) {
  int i = blockIdx.x * blockDim.x + threadIdx.x;
  if (i < total) a[i] = fmaxf(a[i] + b[i & 127], 0.0f);
}

// ---------------- edge scatter: one wave per (edge or self-loop) ----------------
__global__ void edge_scatter(const float* __restrict__ h, const int* __restrict__ ei,
                             const float* __restrict__ dinv, float* __restrict__ agg) {
  int wave = (int)((blockIdx.x * blockDim.x + threadIdx.x) >> 5);
  int lane = threadIdx.x & 31;
  if (wave >= TOT_EDGES) return;
  int row, col;
  if (wave < N_EDGES) { row = ei[wave]; col = ei[N_EDGES + wave]; }
  else               { row = col = wave - N_EDGES; }              // self-loop
  float norm = dinv[row] * dinv[col];
  float4 v = ((const float4*)(h + (size_t)row * 128))[lane];      // 32 lanes x float4 = 128
  float* dst = agg + (size_t)col * 128 + lane * 4;
  atomicAdd(dst + 0, v.x * norm);
  atomicAdd(dst + 1, v.y * norm);
  atomicAdd(dst + 2, v.z * norm);
  atomicAdd(dst + 3, v.w * norm);
}

// ---------------- pooling ----------------
__global__ void pool_scatter(const float* __restrict__ h, const int* __restrict__ batch,
                             float* __restrict__ pooled, float* __restrict__ cnt) {
  int node = (int)((blockIdx.x * blockDim.x + threadIdx.x) >> 5);
  int lane = threadIdx.x & 31;
  if (node >= N_NODES) return;
  int g = batch[node];
  float4 v = ((const float4*)(h + (size_t)node * 128))[lane];
  float* dst = pooled + (size_t)g * 128 + lane * 4;
  atomicAdd(dst + 0, v.x); atomicAdd(dst + 1, v.y);
  atomicAdd(dst + 2, v.z); atomicAdd(dst + 3, v.w);
  if (lane == 0) atomicAdd(&cnt[g], 1.0f);
}
__global__ void pool_finalize(float* __restrict__ pooled, const float* __restrict__ cnt,
                              float* __restrict__ emb_out) {
  int i = blockIdx.x * blockDim.x + threadIdx.x;
  if (i >= N_GRAPHS * HIDDEN) return;
  float c = cnt[i >> 7];
  c = (c > 1.0f) ? c : 1.0f;
  float v = pooled[i] / c;
  pooled[i] = v;
  emb_out[i] = v;
}
__global__ void logits_kernel(const float* __restrict__ pooled, const float* __restrict__ Wlin,
                              const float* __restrict__ blin, float* __restrict__ out) {
  int i = blockIdx.x * blockDim.x + threadIdx.x;
  if (i >= N_GRAPHS * N_CLASSES) return;
  int g = i / N_CLASSES, c = i % N_CLASSES;
  float s = blin[c];
  const float* p = pooled + (size_t)g * 128;
#pragma unroll 8
  for (int j = 0; j < 128; ++j) s += p[j] * Wlin[j * N_CLASSES + c];
  out[i] = s;
}

// ---------------- driver ----------------
extern "C" void kernel_launch(void* const* d_in, const int* in_sizes, int n_in,
                              void* d_out, int out_size, void* d_ws, size_t ws_size,
                              hipStream_t stream) {
  (void)in_sizes; (void)n_in; (void)out_size; (void)ws_size;
  const float* x     = (const float*)d_in[0];
  const int*   ei    = (const int*)d_in[1];   // [2, E] flattened (harness uploads ints as i32)
  const int*   batch = (const int*)d_in[2];
  const float* Wc[6] = { (const float*)d_in[3], (const float*)d_in[5], (const float*)d_in[7],
                         (const float*)d_in[9], (const float*)d_in[11], (const float*)d_in[13] };
  const float* Bc[6] = { (const float*)d_in[4], (const float*)d_in[6], (const float*)d_in[8],
                         (const float*)d_in[10], (const float*)d_in[12], (const float*)d_in[14] };
  const float* Wlin  = (const float*)d_in[15];
  const float* blin  = (const float*)d_in[16];
  float* out = (float*)d_out;  // [0,1280): logits, [1280, 1280+16384): embeddings

  char* ws = (char*)d_ws;
  float* h      = (float*)(ws);                               // 25.6 MB
  float* agg    = (float*)(ws + (size_t)FEAT_TOT * 4);        // 25.6 MB
  float* dinv   = (float*)(ws + (size_t)FEAT_TOT * 8);        // 200 KB
  float* pooled = (float*)(ws + (size_t)FEAT_TOT * 8 + 204800);
  float* cnt    = (float*)(ws + (size_t)FEAT_TOT * 8 + 204800 + N_GRAPHS * HIDDEN * 4);

  const int T = 256;
  const int nbNodes = (N_NODES + T - 1) / T;
  const int nbEdges = (N_EDGES + T - 1) / T;
  const int nbFeat  = (FEAT_TOT + T - 1) / T;
  const int nbGemm  = ((N_NODES / 16) + 7) / 8;           // 3125 row tiles, 8 waves/block
  const int nbScat  = (TOT_EDGES * 32 + T - 1) / T;       // one wave per edge
  const int nbPool  = (N_NODES * 32 + T - 1) / T;         // one wave per node

  // symmetric deg^-1/2 with self-loops
  fill_f32<<<nbNodes, T, 0, stream>>>(dinv, 1.0f, N_NODES);      // self-loop contributes 1
  deg_count<<<nbEdges, T, 0, stream>>>(ei, dinv);
  deg_to_dinv<<<nbNodes, T, 0, stream>>>(dinv, N_NODES);

  // 5 GCN layers
  const float* cur = x;
  for (int l = 0; l < 5; ++l) {
    gemm_bf16_wmma<<<nbGemm, T, 0, stream>>>(cur, Wc[l], h, N_NODES);
    bias_broadcast<<<nbFeat, T, 0, stream>>>(agg, Bc[l], FEAT_TOT);
    edge_scatter<<<nbScat, T, 0, stream>>>(h, ei, dinv, agg);
    relu_inplace<<<nbFeat, T, 0, stream>>>(agg, FEAT_TOT);
    cur = agg;
  }

  // FC + ReLU
  gemm_bf16_wmma<<<nbGemm, T, 0, stream>>>(agg, Wc[5], h, N_NODES);
  bias_relu_inplace<<<nbFeat, T, 0, stream>>>(h, Bc[5], FEAT_TOT);

  // global mean pool + head
  fill_f32<<<(N_GRAPHS * HIDDEN + N_GRAPHS + T - 1) / T, T, 0, stream>>>(
      pooled, 0.0f, N_GRAPHS * HIDDEN + N_GRAPHS);  // pooled and cnt are contiguous
  pool_scatter<<<nbPool, T, 0, stream>>>(h, batch, pooled, cnt);
  pool_finalize<<<(N_GRAPHS * HIDDEN + T - 1) / T, T, 0, stream>>>(pooled, cnt,
                                                                   out + N_GRAPHS * N_CLASSES);
  logits_kernel<<<(N_GRAPHS * N_CLASSES + T - 1) / T, T, 0, stream>>>(pooled, Wlin, blin, out);
}
